// BiasedMultiHeadAttention_42339787604841
// MI455X (gfx1250) — matrix operand order repaired and checked by reading, not compiled, not run on hardware
//
#include <hip/hip_runtime.h>
#include <hip/hip_bf16.h>

// Problem constants (from the reference).
#define B_  4
#define H_  16
#define T_  1024
#define S_  2048
#define HD_ 64
#define E_  1024

#define LOG2E 1.44269504088896340736f

typedef __attribute__((ext_vector_type(16))) _Float16 v16h;
typedef __attribute__((ext_vector_type(8)))  _Float16 v8h;
typedef __attribute__((ext_vector_type(8)))  float    v8f;
typedef __attribute__((ext_vector_type(4)))  float    v4f;

__device__ __forceinline__ v8f wmma_f16(v16h a, v16h b, v8f c) {
  // (neg_a, A, neg_b, B, c_mod, C, reuse_a, reuse_b) -> v_wmma_f32_16x16x32_f16
  return __builtin_amdgcn_wmma_f32_16x16x32_f16(false, a, false, b, (short)0, c,
                                                false, false);
}

__device__ __forceinline__ v16h cat16(v8h lo, v8h hi) {
  v16h r;
#pragma unroll
  for (int j = 0; j < 8; ++j) { r[j] = lo[j]; r[8 + j] = hi[j]; }
  return r;
}

__device__ __forceinline__ v8f zero8() {
  v8f r;
#pragma unroll
  for (int j = 0; j < 8; ++j) r[j] = 0.f;
  return r;
}

__device__ __forceinline__ v8h cvt8(v4f a, v4f b) {
  v8h r;
#pragma unroll
  for (int j = 0; j < 4; ++j) { r[j] = (_Float16)a[j]; r[4 + j] = (_Float16)b[j]; }
  return r;
}

// ---------------------------------------------------------------------------
// Projection GEMM: C[M,N] = (A @ W^T + bvec) * scale, K = N = 1024.
// mode 0: A=f32, out f16 as Q  -> [B,H,T,hd]   (scale = hd^-0.5 * log2e)
// mode 1: A=f32, out f16 as K  -> [B,H,S,hd]
// mode 2: A=f32, out f16 as V^T-> [B,H,hd,S]
// mode 3: A=f16, out f32 row-major [M, E]  (O projection)
// Block = 256 threads (8 waves), block tile 128x128, k-step 32,
// double-buffered LDS (2 x 16 KB). Pipeline order per step:
//   barrier -> issue next-tile global loads -> 8 WMMAs -> cvt+ds_store,
// so the global loads fly across the whole compute phase and the barrier's
// implicit full fence never waits on an in-flight prefetch.
// ---------------------------------------------------------------------------
__global__ __launch_bounds__(256) void proj_gemm_kernel(
    const float* __restrict__ Af, const _Float16* __restrict__ Ah,
    const float* __restrict__ W, const float* __restrict__ bvec,
    _Float16* __restrict__ outh, float* __restrict__ outf,
    int rowsPerB, float scale, int mode)
{
  const int m0   = blockIdx.y * 128;
  const int n0   = blockIdx.x * 128;
  const int tid  = threadIdx.x;
  const int wave = tid >> 5, lane = tid & 31;
  const int ln16 = lane & 15, lh = lane >> 4;
  const int wm   = wave >> 2, wn = wave & 3;

  __shared__ alignas(32) _Float16 As[2][128 * 32];  // [m][k] f16
  __shared__ alignas(32) _Float16 Bs[2][128 * 32];  // [n][k] f16 (W tile direct)

  v8f acc[4][2];
#pragma unroll
  for (int mi = 0; mi < 4; ++mi)
#pragma unroll
    for (int ni = 0; ni < 2; ++ni) acc[mi][ni] = zero8();

  const int ldr = tid >> 1;        // staging row 0..127
  const int ldc = (tid & 1) * 16;  // staging col 0 or 16 (16 elems each)

  // Register staging for the prefetched (next) k-tile.
  v4f aRegF[4];
  v8h aRegH[2];
  v4f wReg[4];

  auto prefetch_tiles = [&](int k0) {
    if (mode < 3) {
      const float* srcA = Af + (size_t)(m0 + ldr) * E_ + k0 + ldc;
#pragma unroll
      for (int q = 0; q < 4; ++q) aRegF[q] = ((const v4f*)srcA)[q];
    } else {
      const _Float16* srcA = Ah + (size_t)(m0 + ldr) * E_ + k0 + ldc;
      aRegH[0] = ((const v8h*)srcA)[0];
      aRegH[1] = ((const v8h*)srcA)[1];
    }
    const float* srcW = W + (size_t)(n0 + ldr) * E_ + k0 + ldc;
#pragma unroll
    for (int q = 0; q < 4; ++q) wReg[q] = ((const v4f*)srcW)[q];
  };

  auto commit_tiles = [&](int buf) {
    v8h a0, a1;
    if (mode < 3) {
      a0 = cvt8(aRegF[0], aRegF[1]);
      a1 = cvt8(aRegF[2], aRegF[3]);
    } else {
      a0 = aRegH[0];
      a1 = aRegH[1];
    }
    *(v8h*)&As[buf][ldr * 32 + ldc]     = a0;
    *(v8h*)&As[buf][ldr * 32 + ldc + 8] = a1;
    *(v8h*)&Bs[buf][ldr * 32 + ldc]     = cvt8(wReg[0], wReg[1]);
    *(v8h*)&Bs[buf][ldr * 32 + ldc + 8] = cvt8(wReg[2], wReg[3]);
  };

  auto compute_tile = [&](int buf) {
    v16h bf[2];
#pragma unroll
    for (int ni = 0; ni < 2; ++ni)
      bf[ni] = *(const v16h*)&Bs[buf][(wn * 32 + ni * 16 + ln16) * 32 + lh * 16];
#pragma unroll
    for (int mi = 0; mi < 4; ++mi) {
      const _Float16* ar = &As[buf][(wm * 64 + mi * 16 + ln16) * 32];
      const int g = lh * 16;
      v16h af = *(const v16h*)(ar + g);
#pragma unroll
      for (int ni = 0; ni < 2; ++ni)
        acc[mi][ni] = wmma_f16(af, bf[ni], acc[mi][ni]);
    }
  };

  constexpr int NSTEP = E_ / 32;
  prefetch_tiles(0);
  commit_tiles(0);

  for (int step = 0; step < NSTEP - 1; ++step) {
    const int buf = step & 1;
    __syncthreads();                       // LDS[buf] visible (loadcnt is 0 here)
    prefetch_tiles((step + 1) * 32);       // issue loads AFTER the fence
    compute_tile(buf);                     // 8 WMMAs while loads are in flight
    commit_tiles(buf ^ 1);                 // cvt waits loads; ds_store next buf
  }
  __syncthreads();
  compute_tile((NSTEP - 1) & 1);

  // --- epilogue: +bias, *scale, layout-specific store ---
#pragma unroll
  for (int ni = 0; ni < 2; ++ni) {
    const int gcol = n0 + wn * 32 + ni * 16 + ln16;
    const float bv = bvec[gcol];
#pragma unroll
    for (int mi = 0; mi < 4; ++mi) {
#pragma unroll
      for (int i = 0; i < 8; ++i) {
        const int grow = m0 + wm * 64 + mi * 16 + i + 8 * lh;
        const float v = (acc[mi][ni][i] + bv) * scale;
        if (mode == 3) {
          outf[(size_t)grow * E_ + gcol] = v;
        } else {
          const int b = grow / rowsPerB, r = grow - b * rowsPerB;
          const int h = gcol >> 6, d = gcol & 63;
          size_t idx;
          if (mode == 2)  // V transposed: [B,H,hd,S]
            idx = (((size_t)(b * H_ + h)) * HD_ + d) * rowsPerB + r;
          else            // Q/K: [B,H,rows,hd]
            idx = (((size_t)(b * H_ + h)) * rowsPerB + r) * HD_ + d;
          outh[idx] = (_Float16)v;
        }
      }
    }
  }
}

// ---------------------------------------------------------------------------
// Fused biased flash attention.
// One wave = 16 query rows; S streamed in chunks of 32.
// grid = (B*H, T/128), block = 256 (8 waves).
// Q pre-scaled by hd^-0.5 * log2e; bias scaled by log2e at add time, so the
// softmax uses exp2 (the native v_exp_f32 semantics) directly.
// K/V (32 MB) are re-read by the 8 T-blocks per head but live in the 192 MB
// L2; bias (512 MB) is streamed exactly once, with a 2-chunk-ahead prefetch.
// ---------------------------------------------------------------------------
__global__ __launch_bounds__(256) void flash_attn_kernel(
    const _Float16* __restrict__ Qh,   // [B,H,T,hd]
    const _Float16* __restrict__ Kh,   // [B,H,S,hd]
    const _Float16* __restrict__ Vt,   // [B,H,hd,S]
    const float*    __restrict__ bias, // [B,H,T,S]
    _Float16*       __restrict__ attn) // [B,H,T,hd]
{
  const int bh   = blockIdx.x;
  const int wave = threadIdx.x >> 5, lane = threadIdx.x & 31;
  const int ln16 = lane & 15, lh = lane >> 4;
  const int t0   = blockIdx.y * 128 + wave * 16;

  const _Float16* Qp = Qh + ((size_t)bh * T_ + t0) * HD_;
  const _Float16* Kp = Kh + (size_t)bh * S_ * HD_;
  const _Float16* Vp = Vt + (size_t)bh * HD_ * S_;
  const float*    Bp = bias + ((size_t)bh * T_ + t0) * S_;
  _Float16*       Op = attn + ((size_t)bh * T_ + t0) * HD_;

  // Per-wave P staging slab (C-frag layout -> A-frag layout via LDS).
  __shared__ alignas(32) _Float16 Pl[8][16 * 32];
  _Float16* myP = &Pl[wave][0];

  // Q A-fragments (hd=64 -> two k=32 steps), resident for the whole S loop.
  v16h qa[2];
  {
    const _Float16* qr = Qp + ln16 * HD_;
    const int g = lh * 16;
#pragma unroll
    for (int ks = 0; ks < 2; ++ks)
      qa[ks] = *(const v16h*)(qr + ks * 32 + g);
  }

  v8f acc[4];   // O accumulator 16x64 (four 16x16 C frags)
#pragma unroll
  for (int nt = 0; nt < 4; ++nt) acc[nt] = zero8();
  float mr[8], lr[8];  // per-row running max / sum (row = i + 8*lh)
#pragma unroll
  for (int i = 0; i < 8; ++i) { mr[i] = -1e30f; lr[i] = 0.f; }

  for (int s0 = 0; s0 < S_; s0 += 32) {
    // ---- prefetch bias two chunks ahead (streamed-once 512 MB tensor) ----
    if (s0 + 64 < S_) {
      // 32 lanes cover 16 rows x two 64B half-chunks.
      __builtin_prefetch(Bp + (size_t)ln16 * S_ + s0 + 64 + lh * 16, 0, 1);
    }

    // ---- scores S16x32 = Q(16x64) @ K^T ----
    v8f sc[2];
#pragma unroll
    for (int sj = 0; sj < 2; ++sj) {
      sc[sj] = zero8();
#pragma unroll
      for (int ks = 0; ks < 2; ++ks) {
        v16h kb = *(const v16h*)(Kp + (size_t)(s0 + sj * 16 + ln16) * HD_ +
                                 ks * 32 + lh * 16);
        sc[sj] = wmma_f16(qa[ks], kb, sc[sj]);
      }
    }
    // ---- + bias * log2e (single pass over the 512MB tensor) ----
#pragma unroll
    for (int sj = 0; sj < 2; ++sj)
#pragma unroll
      for (int i = 0; i < 8; ++i)
        sc[sj][i] += Bp[(size_t)(i + 8 * lh) * S_ + s0 + sj * 16 + ln16] * LOG2E;

    // ---- online softmax in log2 domain (rows live across a 16-lane group) --
#pragma unroll
    for (int i = 0; i < 8; ++i) {
      float v = fmaxf(sc[0][i], sc[1][i]);
      v = fmaxf(v, __shfl_xor(v, 1));
      v = fmaxf(v, __shfl_xor(v, 2));
      v = fmaxf(v, __shfl_xor(v, 4));
      v = fmaxf(v, __shfl_xor(v, 8));
      const float nm    = fmaxf(mr[i], v);
      const float alpha = exp2f(mr[i] - nm);
      mr[i] = nm;
      const float p0 = exp2f(sc[0][i] - nm);
      const float p1 = exp2f(sc[1][i] - nm);
      sc[0][i] = p0; sc[1][i] = p1;
      float rs = p0 + p1;
      rs += __shfl_xor(rs, 1);
      rs += __shfl_xor(rs, 2);
      rs += __shfl_xor(rs, 4);
      rs += __shfl_xor(rs, 8);
      lr[i] = lr[i] * alpha + rs;
#pragma unroll
      for (int nt = 0; nt < 4; ++nt) acc[nt][i] *= alpha;
    }

    // ---- P: C-frag f32 -> LDS f16 -> A-frag (wave-private, DS in-order) ----
#pragma unroll
    for (int sj = 0; sj < 2; ++sj)
#pragma unroll
      for (int i = 0; i < 8; ++i)
        myP[(i + 8 * lh) * 32 + sj * 16 + ln16] = (_Float16)sc[sj][i];

    v16h pa;
    {
      const _Float16* pr = myP + ln16 * 32;
      const int g = lh * 16;
      pa = *(const v16h*)(pr + g);
    }

    // ---- O += P(16x32) @ V(32x64); V stored transposed => contiguous B frags
#pragma unroll
    for (int nt = 0; nt < 4; ++nt) {
      v16h vb = *(const v16h*)(Vp + (size_t)(nt * 16 + ln16) * S_ + s0 + lh * 16);
      acc[nt] = wmma_f16(pa, vb, acc[nt]);
    }
  }

  // ---- normalize + store f16 [B,H,T,hd] (reference's flat reshape order) ----
  float inv[8];
#pragma unroll
  for (int i = 0; i < 8; ++i) inv[i] = 1.f / lr[i];
#pragma unroll
  for (int nt = 0; nt < 4; ++nt)
#pragma unroll
    for (int i = 0; i < 8; ++i)
      Op[(size_t)(i + 8 * lh) * HD_ + nt * 16 + ln16] =
          (_Float16)(acc[nt][i] * inv[i]);
}

// ---------------------------------------------------------------------------
extern "C" void kernel_launch(void* const* d_in, const int* in_sizes, int n_in,
                              void* d_out, int out_size, void* d_ws, size_t ws_size,
                              hipStream_t stream) {
  const float* hidden = (const float*)d_in[0];  // [B,T,E]
  const float* kv     = (const float*)d_in[1];  // [B,S,E]
  const float* bias   = (const float*)d_in[2];  // [B,H,T,S]
  const float* Wq = (const float*)d_in[3];
  const float* bq = (const float*)d_in[4];
  const float* Wk = (const float*)d_in[5];
  const float* bk = (const float*)d_in[6];
  const float* Wv = (const float*)d_in[7];
  const float* bv = (const float*)d_in[8];
  const float* Wo = (const float*)d_in[9];
  const float* bo = (const float*)d_in[10];
  float* out = (float*)d_out;

  // Workspace: f16 intermediates, 48 MB total.
  const size_t QN = (size_t)B_ * H_ * T_ * HD_;  // 4M elems
  const size_t KN = (size_t)B_ * H_ * S_ * HD_;  // 8M elems
  _Float16* Qh = (_Float16*)d_ws;   // [B,H,T,hd]   8 MB
  _Float16* Kh = Qh + QN;           // [B,H,S,hd]  16 MB
  _Float16* Vt = Kh + KN;           // [B,H,hd,S]  16 MB
  _Float16* At = Vt + KN;           // [B,H,T,hd]   8 MB
  (void)ws_size; (void)in_sizes; (void)n_in; (void)out_size;

  const dim3 blk(256);
  const float qscale = 0.125f * LOG2E;  // hd^-0.5, log2e folded for exp2 softmax

  // Q/K/V projections (f32 in, f16 out; V written transposed).
  proj_gemm_kernel<<<dim3(E_ / 128, (B_ * T_) / 128), blk, 0, stream>>>(
      hidden, nullptr, Wq, bq, Qh, nullptr, T_, qscale, 0);
  proj_gemm_kernel<<<dim3(E_ / 128, (B_ * S_) / 128), blk, 0, stream>>>(
      kv, nullptr, Wk, bk, Kh, nullptr, S_, 1.0f, 1);
  proj_gemm_kernel<<<dim3(E_ / 128, (B_ * S_) / 128), blk, 0, stream>>>(
      kv, nullptr, Wv, bv, Vt, nullptr, S_, 1.0f, 2);

  // Fused biased flash attention.
  flash_attn_kernel<<<dim3(B_ * H_, T_ / 128), blk, 0, stream>>>(
      Qh, Kh, Vt, bias, At);

  // Output projection (f16 in, f32 out row-major == reference's flat reshape).
  proj_gemm_kernel<<<dim3(E_ / 128, (B_ * T_) / 128), blk, 0, stream>>>(
      nullptr, At, Wo, bo, nullptr, out, T_, 1.0f, 3);
}
